// MambaUNet_816043786859
// MI455X (gfx1250) — matrix operand order repaired and checked by reading, not compiled, not run on hardware
//
#include <hip/hip_runtime.h>
#include <math.h>

// ---------------------------------------------------------------------------
// Types for CDNA5 WMMA
// ---------------------------------------------------------------------------
typedef __attribute__((ext_vector_type(16))) _Float16 v16h;
typedef __attribute__((ext_vector_type(8)))  _Float16 v8h;
typedef __attribute__((ext_vector_type(8)))  float    v8f;

#define ACT_NONE 0
#define ACT_RELU 1
#define ACT_SILU 2
#define ACT_SOFTPLUS 3
#define ACT_SIGMOID 4

__device__ __forceinline__ float act_apply(float v, int act) {
  switch (act) {
    case ACT_RELU:     return v > 0.f ? v : 0.f;
    case ACT_SILU:     return v / (1.f + __expf(-v));
    case ACT_SOFTPLUS: return (v > 20.f) ? v : log1pf(__expf(v));
    case ACT_SIGMOID:  return 1.f / (1.f + __expf(-v));
    default:           return v;
  }
}

// ---------------------------------------------------------------------------
// Pack fp32 (strided, unpadded) -> f16 (row-major, M padded to 16, K padded to
// 32, zero-filled). Guarantees the WMMA kernel needs no bounds checks on loads
// and that every fragment load is a 16/32-byte aligned vector load.
// ---------------------------------------------------------------------------
__global__ void pack_f16_kernel(const float* __restrict__ src, int lds, int M, int K,
                                _Float16* __restrict__ dst, int Kp, int total)
{
  int idx = blockIdx.x * blockDim.x + threadIdx.x;
  if (idx >= total) return;
  int k = idx % Kp, m = idx / Kp;
  float v = (m < M && k < K) ? src[(size_t)m * lds + k] : 0.f;
  dst[idx] = (_Float16)v;
}

// ---------------------------------------------------------------------------
// WMMA GEMM:  C[M,N] = act( A[M,K] * W[N,K]^T + bias[N] )   (i.e.  x @ W.T + b)
// Operands pre-packed to f16 with K padded to Kp (mult of 32), rows padded to
// tile multiples. One 16x16 C tile per wave32; 4 waves / 128-thread block.
// Inner loop: 4 aligned vector loads + v_wmma_f32_16x16x32_f16, no branches.
// Fragment layouts per CDNA5 ISA 7.12.2.
// ---------------------------------------------------------------------------
__global__ __launch_bounds__(128) void gemm_wmma_kernel(
    const _Float16* __restrict__ AH,     // Mp x Kp
    const _Float16* __restrict__ WH,     // Np x Kp
    const float* __restrict__ bias,      // may be null
    float* __restrict__ C, int ldc,
    int M, int N, int Kp, int act)
{
  const int wave = threadIdx.x >> 5;
  const int lane = threadIdx.x & 31;
  const int lhi  = lane >> 4;            // half-wave index
  const int llo  = lane & 15;
  const int tiles_n = (N + 15) >> 4;
  const int tiles_m = (M + 15) >> 4;
  const int tile = blockIdx.x * 4 + wave;
  if (tile >= tiles_m * tiles_n) return; // wave-uniform: EXEC stays full for WMMA
  const int tm = tile / tiles_n;
  const int tn = tile % tiles_n;
  const int row = tm * 16 + llo;         // A row held by this lane (padded: in range)
  const int col = tn * 16 + llo;         // B column held by this lane (padded: in range)

  const _Float16* ap = AH + (size_t)row * Kp;
  const _Float16* bp = WH + (size_t)col * Kp;

  v8f acc = {};
  for (int kk = 0; kk < Kp; kk += 32) {
    // A 16x32 fragment: lane holds row (lane&15); element e maps to
    // k = 16*(e>>3) + 8*lhi + (e&7)  -> two contiguous 8-half runs
    v8h a0 = *(const v8h*)(ap + kk + (lhi << 3));
    v8h a1 = *(const v8h*)(ap + kk + 16 + (lhi << 3));
    // B 32x16 fragment: lane holds column (lane&15); k = 16*(e>>3) + 8*lhi + (e&7)
    v8h b0 = *(const v8h*)(bp + kk + (lhi << 3));
    v8h b1 = *(const v8h*)(bp + kk + 16 + (lhi << 3));
    v16h bf;
    v16h af;
#pragma unroll
    for (int e = 0; e < 8; ++e) { af[e] = a0[e]; af[e + 8] = a1[e]; bf[e] = b0[e]; bf[e + 8] = b1[e]; }
    if (kk + 32 < Kp) {
      __builtin_prefetch(ap + kk + 32, 0, 0);
      __builtin_prefetch(bp + kk + 32, 0, 0);
    }
    acc = __builtin_amdgcn_wmma_f32_16x16x32_f16(
        /*neg_a=*/false, af, /*neg_b=*/false, bf,
        /*c_mod=*/(short)0, acc, /*reuse_a=*/false, /*reuse_b=*/false);
  }
  // f32 C/D: VGPR r holds row (r + 8*lhi), col = lane&15
  const bool colok = col < N;
#pragma unroll
  for (int r8 = 0; r8 < 8; ++r8) {
    int m = tm * 16 + r8 + lhi * 8;
    if (m < M && colok) {
      float v = acc[r8];
      if (bias) v += bias[col];
      C[(size_t)m * ldc + col] = act_apply(v, act);
    }
  }
}

// ---------------------------------------------------------------------------
// Direct conv2d (NCHW, OIHW weights), fp32
// ---------------------------------------------------------------------------
__global__ void conv2d_kernel(const float* __restrict__ x, const float* __restrict__ w,
                              const float* __restrict__ b, float* __restrict__ y,
                              int Cin, int Cout, int H, int W, int KH, int KW, int pad)
{
  int idx = blockIdx.x * blockDim.x + threadIdx.x;
  int total = Cout * H * W;
  if (idx >= total) return;
  int ow = idx % W, oh = (idx / W) % H, oc = idx / (W * H);
  float s = b ? b[oc] : 0.f;
  for (int ic = 0; ic < Cin; ++ic) {
    const float* xp = x + (size_t)ic * H * W;
    const float* wp = w + ((size_t)oc * Cin + ic) * KH * KW;
    for (int kh = 0; kh < KH; ++kh) {
      int ih = oh + kh - pad;
      if (ih < 0 || ih >= H) continue;
      for (int kw = 0; kw < KW; ++kw) {
        int iw = ow + kw - pad;
        if (iw < 0 || iw >= W) continue;
        s += xp[ih * W + iw] * wp[kh * KW + kw];
      }
    }
  }
  y[idx] = s;
}

// ---------------------------------------------------------------------------
// BatchNorm (training mode, batch stats) : stats then apply(+optional relu)
// ---------------------------------------------------------------------------
__global__ __launch_bounds__(256) void bn_stats_kernel(const float* __restrict__ x,
                                                       float* __restrict__ mv, int HW)
{
  __shared__ float ss[256], sq[256];
  int c = blockIdx.x;
  const float* xp = x + (size_t)c * HW;
  float s = 0.f, q = 0.f;
  for (int i = threadIdx.x; i < HW; i += 256) { float v = xp[i]; s += v; q += v * v; }
  ss[threadIdx.x] = s; sq[threadIdx.x] = q;
  __syncthreads();
  for (int o = 128; o > 0; o >>= 1) {
    if ((int)threadIdx.x < o) { ss[threadIdx.x] += ss[threadIdx.x + o]; sq[threadIdx.x] += sq[threadIdx.x + o]; }
    __syncthreads();
  }
  if (threadIdx.x == 0) {
    float m = ss[0] / HW;
    mv[2 * c] = m;
    mv[2 * c + 1] = sq[0] / HW - m * m;
  }
}

__global__ void bn_apply_kernel(float* __restrict__ x, const float* __restrict__ mv,
                                const float* __restrict__ g, const float* __restrict__ b,
                                int C, int HW, int relu)
{
  int idx = blockIdx.x * blockDim.x + threadIdx.x;
  if (idx >= C * HW) return;
  int c = idx / HW;
  float o = (x[idx] - mv[2 * c]) * rsqrtf(mv[2 * c + 1] + 1e-5f) * g[c] + b[c];
  x[idx] = relu ? fmaxf(o, 0.f) : o;
}

// ---------------------------------------------------------------------------
// Misc elementwise / reshape / pooling kernels
// ---------------------------------------------------------------------------
__global__ void maxpool2_kernel(const float* __restrict__ x, float* __restrict__ y,
                                int C, int H, int W)
{
  int Ho = H >> 1, Wo = W >> 1;
  int idx = blockIdx.x * blockDim.x + threadIdx.x;
  if (idx >= C * Ho * Wo) return;
  int ow = idx % Wo, oh = (idx / Wo) % Ho, c = idx / (Wo * Ho);
  const float* xp = x + ((size_t)c * H + 2 * oh) * W + 2 * ow;
  y[idx] = fmaxf(fmaxf(xp[0], xp[1]), fmaxf(xp[W], xp[W + 1]));
}

__global__ void chw2lc_kernel(const float* __restrict__ x, float* __restrict__ s, int C, int L)
{
  int idx = blockIdx.x * blockDim.x + threadIdx.x;
  if (idx >= C * L) return;
  int c = idx % C, l = idx / C;
  s[idx] = x[(size_t)c * L + l];
}

__global__ void lc2chw_kernel(const float* __restrict__ s, float* __restrict__ x, int C, int L)
{
  int idx = blockIdx.x * blockDim.x + threadIdx.x;
  if (idx >= C * L) return;
  int c = idx % C, l = idx / C;
  x[(size_t)c * L + l] = s[idx];
}

// depthwise causal conv1d (k=4, left pad 3) + bias + SiLU; input is xz (L x 2di), first di chans
__global__ void dwconv_silu_kernel(const float* __restrict__ xz, const float* __restrict__ w,
                                   const float* __restrict__ b, float* __restrict__ u,
                                   int L, int di)
{
  int idx = blockIdx.x * blockDim.x + threadIdx.x;
  if (idx >= L * di) return;
  int d = idx % di, l = idx / di;
  float s = b[d];
#pragma unroll
  for (int j = 0; j < 4; ++j) {
    int ll = l - 3 + j;
    if (ll >= 0) s += xz[(size_t)ll * 2 * di + d] * w[d * 4 + j];
  }
  u[idx] = s / (1.f + __expf(-s));
}

__global__ void negexp_kernel(const float* __restrict__ a, float* __restrict__ o, int n)
{
  int i = blockIdx.x * blockDim.x + threadIdx.x;
  if (i < n) o[i] = -__expf(a[i]);
}

// Selective scan: half-wave-of-16 lanes = 16 states of one channel d.
// h[l] = exp(dt*A)*h[l-1] + dt*u*B ; y[l,d] = sum_n h*C + u*D  (shfl reduction)
__global__ __launch_bounds__(128) void scan_kernel(
    const float* __restrict__ u, const float* __restrict__ dt,
    const float* __restrict__ An, const float* __restrict__ dbl,
    const float* __restrict__ Dv, float* __restrict__ y,
    int L, int di, int r)
{
  int ln = threadIdx.x & 15;
  int d = blockIdx.x * 8 + (threadIdx.x >> 4);
  if (d >= di) return;
  const int rw = r + 32;
  float a  = An[d * 16 + ln];
  float Dd = Dv[d];
  float h = 0.f;
  for (int l = 0; l < L; ++l) {
    float dtv = dt[(size_t)l * di + d];
    float uv  = u[(size_t)l * di + d];
    float Bv  = dbl[(size_t)l * rw + r + ln];
    float Cv  = dbl[(size_t)l * rw + r + 16 + ln];
    h = __expf(dtv * a) * h + dtv * uv * Bv;
    float part = h * Cv;
    part += __shfl_xor(part, 1, 32);
    part += __shfl_xor(part, 2, 32);
    part += __shfl_xor(part, 4, 32);
    part += __shfl_xor(part, 8, 32);
    if (ln == 0) y[(size_t)l * di + d] = part + uv * Dd;
  }
}

__global__ void gate_kernel(float* __restrict__ y, const float* __restrict__ xz, int L, int di)
{
  int idx = blockIdx.x * blockDim.x + threadIdx.x;
  if (idx >= L * di) return;
  int d = idx % di, l = idx / di;
  float z = xz[(size_t)l * 2 * di + di + d];
  y[idx] *= z / (1.f + __expf(-z));
}

__global__ void layernorm_kernel(float* __restrict__ x, const float* __restrict__ g,
                                 const float* __restrict__ b, int L, int C)
{
  int l = blockIdx.x * blockDim.x + threadIdx.x;
  if (l >= L) return;
  float* row = x + (size_t)l * C;
  float m = 0.f;
  for (int c = 0; c < C; ++c) m += row[c];
  m /= C;
  float v = 0.f;
  for (int c = 0; c < C; ++c) { float d = row[c] - m; v += d * d; }
  v /= C;
  float inv = rsqrtf(v + 1e-5f);
  for (int c = 0; c < C; ++c) row[c] = (row[c] - m) * inv * g[c] + b[c];
}

__global__ __launch_bounds__(256) void spatial_mean_kernel(const float* __restrict__ x,
                                                           float* __restrict__ out, int HW)
{
  __shared__ float ss[256];
  int c = blockIdx.x;
  float s = 0.f;
  for (int i = threadIdx.x; i < HW; i += 256) s += x[(size_t)c * HW + i];
  ss[threadIdx.x] = s; __syncthreads();
  for (int o = 128; o > 0; o >>= 1) {
    if ((int)threadIdx.x < o) ss[threadIdx.x] += ss[threadIdx.x + o];
    __syncthreads();
  }
  if (threadIdx.x == 0) out[c] = ss[0] / HW;
}

__global__ __launch_bounds__(256) void spatial_max_kernel(const float* __restrict__ x,
                                                          float* __restrict__ out, int HW)
{
  __shared__ float ss[256];
  int c = blockIdx.x;
  float s = -3.402823e38f;
  for (int i = threadIdx.x; i < HW; i += 256) s = fmaxf(s, x[(size_t)c * HW + i]);
  ss[threadIdx.x] = s; __syncthreads();
  for (int o = 128; o > 0; o >>= 1) {
    if ((int)threadIdx.x < o) ss[threadIdx.x] = fmaxf(ss[threadIdx.x], ss[threadIdx.x + o]);
    __syncthreads();
  }
  if (threadIdx.x == 0) out[c] = ss[0];
}

__global__ void se_scale_add_kernel(const float* __restrict__ y, const float* __restrict__ se,
                                    const float* __restrict__ res, float* __restrict__ out,
                                    int C, int HW)
{
  int idx = blockIdx.x * blockDim.x + threadIdx.x;
  if (idx >= C * HW) return;
  int c = idx / HW;
  out[idx] = y[idx] * se[c] + res[idx];
}

__global__ void ca_scale_kernel(float* __restrict__ x, const float* __restrict__ ao,
                                const float* __restrict__ mo, int C, int HW)
{
  int idx = blockIdx.x * blockDim.x + threadIdx.x;
  if (idx >= C * HW) return;
  int c = idx / HW;
  float g = ao[c] + mo[c];
  x[idx] *= 1.f / (1.f + __expf(-g));
}

__global__ void chan_meanmax_kernel(const float* __restrict__ x, float* __restrict__ out,
                                    int C, int HW)
{
  int hw = blockIdx.x * blockDim.x + threadIdx.x;
  if (hw >= HW) return;
  float s = 0.f, m = -3.402823e38f;
  for (int c = 0; c < C; ++c) {
    float v = x[(size_t)c * HW + hw];
    s += v; m = fmaxf(m, v);
  }
  out[hw] = s / C;
  out[HW + hw] = m;
}

__global__ void sa_scale_kernel(float* __restrict__ x, const float* __restrict__ sa, int C, int HW)
{
  int idx = blockIdx.x * blockDim.x + threadIdx.x;
  if (idx >= C * HW) return;
  int hw = idx % HW;
  x[idx] *= 1.f / (1.f + __expf(-sa[hw]));
}

// conv-transpose 2x2 stride 2 ; w: (Cin, Cout, 2, 2)
__global__ void convt2x2_kernel(const float* __restrict__ x, const float* __restrict__ w,
                                const float* __restrict__ b, float* __restrict__ y,
                                int Cin, int Cout, int H, int W)
{
  int Ho = 2 * H, Wo = 2 * W;
  int idx = blockIdx.x * blockDim.x + threadIdx.x;
  if (idx >= Cout * Ho * Wo) return;
  int ow = idx % Wo, oh = (idx / Wo) % Ho, oc = idx / (Wo * Ho);
  int h = oh >> 1, k = oh & 1, ww = ow >> 1, l = ow & 1;
  float s = b[oc];
  for (int i = 0; i < Cin; ++i)
    s += x[((size_t)i * H + h) * W + ww] * w[(((size_t)i * Cout + oc) * 2 + k) * 2 + l];
  y[idx] = s;
}

// ---------------------------------------------------------------------------
// Host-side orchestration
// ---------------------------------------------------------------------------
static inline int ceil_div(int a, int b) { return (a + b - 1) / b; }
static inline const float* FP(void* const* d_in, int i) { return (const float*)d_in[i]; }

// Workspace layout (float offsets)
static const size_t OFF_SKIP1 = 0;                       // 64*128*128
static const size_t OFF_SKIP2 = 1048576;                 // 128*64*64
static const size_t OFF_SKIP3 = 1572864;                 // 256*32*32
static const size_t OFF_SKIP4 = 1835008;                 // 512*16*16
static const size_t OFF_BUFA  = 1966080;                 // 2359296
static const size_t OFF_BUFB  = OFF_BUFA + 2359296;      // 2359296
static const size_t OFF_SIN   = OFF_BUFB + 2359296;      // 1048576  (L x c)
static const size_t OFF_XZ    = OFF_SIN + 1048576;       // 4194304  (L x 2di)
static const size_t OFF_U     = OFF_XZ + 4194304;        // 2097152  (L x di)
static const size_t OFF_DBL   = OFF_U + 2097152;         // 1048576  (L x (r+32))
static const size_t OFF_DT    = OFF_DBL + 1048576;       // 2097152  (L x di)
static const size_t OFF_Y     = OFF_DT + 2097152;        // 2097152  (L x di / y_chw)
static const size_t OFF_MOUT  = OFF_Y + 2097152;         // 1048576  (L x c)
static const size_t OFF_ANEG  = OFF_MOUT + 1048576;      // 32768
static const size_t OFF_STAT  = OFF_ANEG + 32768;        // 16384
static const size_t OFF_MV    = OFF_STAT + 16384;        // 4096
static const size_t OFF_AH    = OFF_MV + 4096;           // 1114112 floats = 2.2M f16
static const size_t OFF_WH    = OFF_AH + 1114112;        // 2101248 floats = 4.2M f16

// Param leaf indices: jax-style sorted-key flatten of {'params', 'x'}
enum {
  B_BOT = 0, B_DEC1 = 25, B_DEC2 = 44, B_DEC3 = 63, B_DEC4 = 82,
  B_ENC1 = 101, B_ENC2 = 118, B_ENC3 = 135, B_ENC4 = 152,
  I_FINAL_B = 169, I_FINAL_W = 170, I_X = 171
};
// mamba sub-dict (sorted): A_log,D,conv_b,conv_w,dt_b,dt_w,in_proj,ln_b,ln_g,out_w,se_w1,se_w2,x_proj
struct MP {
  const float *A_log, *D, *conv_b, *conv_w, *dt_b, *dt_w, *in_proj,
              *ln_b, *ln_g, *out_w, *se_w1, *se_w2, *x_proj;
};
static MP mp_from(void* const* d, int base) {
  MP p;
  p.A_log = FP(d, base + 0);  p.D = FP(d, base + 1);     p.conv_b = FP(d, base + 2);
  p.conv_w = FP(d, base + 3); p.dt_b = FP(d, base + 4);  p.dt_w = FP(d, base + 5);
  p.in_proj = FP(d, base + 6); p.ln_b = FP(d, base + 7); p.ln_g = FP(d, base + 8);
  p.out_w = FP(d, base + 9);  p.se_w1 = FP(d, base + 10); p.se_w2 = FP(d, base + 11);
  p.x_proj = FP(d, base + 12);
  return p;
}

static void launch_gemm(hipStream_t st, float* ws, const float* A, int lda, const float* Wt,
                        const float* bias, float* C, int ldc, int M, int N, int K, int act)
{
  int Kp = ceil_div(K, 32) * 32;
  int Mp = ceil_div(M, 16) * 16;
  int Np = ceil_div(N, 16) * 16;
  _Float16* AH = (_Float16*)(ws + OFF_AH);
  _Float16* WH = (_Float16*)(ws + OFF_WH);
  int ta = Mp * Kp, tw = Np * Kp;
  pack_f16_kernel<<<ceil_div(ta, 256), 256, 0, st>>>(A, lda, M, K, AH, Kp, ta);
  pack_f16_kernel<<<ceil_div(tw, 256), 256, 0, st>>>(Wt, K, N, K, WH, Kp, tw);
  int tiles = (Mp / 16) * (Np / 16);
  gemm_wmma_kernel<<<ceil_div(tiles, 4), 128, 0, st>>>(AH, WH, bias, C, ldc, M, N, Kp, act);
}

static void run_bn_relu(hipStream_t st, float* xb, const float* g, const float* b,
                        int C, int HW, float* mv)
{
  bn_stats_kernel<<<C, 256, 0, st>>>(xb, mv, HW);
  bn_apply_kernel<<<ceil_div(C * HW, 256), 256, 0, st>>>(xb, mv, g, b, C, HW, 1);
}

static void run_mamba(hipStream_t st, const MP& p, const float* x_chw, float* out_chw,
                      int c, int H, int W, float* ws)
{
  const int L = H * W, di = 2 * c, r = c / 16, rw = r + 32;
  float* S   = ws + OFF_SIN;
  float* XZ  = ws + OFF_XZ;
  float* U   = ws + OFF_U;
  float* DBL = ws + OFF_DBL;
  float* DT  = ws + OFF_DT;
  float* Y   = ws + OFF_Y;
  float* MO  = ws + OFF_MOUT;
  float* AN  = ws + OFF_ANEG;
  float* ST  = ws + OFF_STAT;

  chw2lc_kernel<<<ceil_div(L * c, 256), 256, 0, st>>>(x_chw, S, c, L);
  launch_gemm(st, ws, S, c, p.in_proj, nullptr, XZ, 2 * di, L, 2 * di, c, ACT_NONE);
  dwconv_silu_kernel<<<ceil_div(L * di, 256), 256, 0, st>>>(XZ, p.conv_w, p.conv_b, U, L, di);
  launch_gemm(st, ws, U, di, p.x_proj, nullptr, DBL, rw, L, rw, di, ACT_NONE);
  launch_gemm(st, ws, DBL, rw, p.dt_w, p.dt_b, DT, di, L, di, r, ACT_SOFTPLUS);
  negexp_kernel<<<ceil_div(di * 16, 256), 256, 0, st>>>(p.A_log, AN, di * 16);
  scan_kernel<<<ceil_div(di, 8), 128, 0, st>>>(U, DT, AN, DBL, p.D, Y, L, di, r);
  gate_kernel<<<ceil_div(L * di, 256), 256, 0, st>>>(Y, XZ, L, di);
  launch_gemm(st, ws, Y, di, p.out_w, nullptr, MO, c, L, c, di, ACT_NONE);
  layernorm_kernel<<<ceil_div(L, 128), 128, 0, st>>>(MO, p.ln_g, p.ln_b, L, c);
  lc2chw_kernel<<<ceil_div(L * c, 256), 256, 0, st>>>(MO, Y, c, L);  // Y reused as y_chw
  spatial_mean_kernel<<<c, 256, 0, st>>>(Y, ST, L);
  launch_gemm(st, ws, ST, c, p.se_w1, nullptr, ST + 2048, c / 16, 1, c / 16, c, ACT_RELU);
  launch_gemm(st, ws, ST + 2048, c / 16, p.se_w2, nullptr, ST + 4096, c, 1, c, c / 16, ACT_SIGMOID);
  se_scale_add_kernel<<<ceil_div(c * L, 256), 256, 0, st>>>(Y, ST + 4096, x_chw, out_chw, c, L);
}

// down block (sorted keys): bn_b+0, bn_g+1, conv_b+2, conv_w+3, mamba+4..16
static void run_down(hipStream_t st, void* const* din, int base, const float* xin,
                     int Cin, int Cout, int H, int W, float* ws, float* skip, float* pooled)
{
  float* feat = ws + OFF_BUFB;
  conv2d_kernel<<<ceil_div(Cout * H * W, 256), 256, 0, st>>>(
      xin, FP(din, base + 3), FP(din, base + 2), feat, Cin, Cout, H, W, 3, 3, 1);
  run_bn_relu(st, feat, FP(din, base + 1), FP(din, base + 0), Cout, H * W, ws + OFF_MV);
  MP p = mp_from(din, base + 4);
  run_mamba(st, p, feat, skip, Cout, H, W, ws);
  maxpool2_kernel<<<ceil_div(Cout * (H / 2) * (W / 2), 256), 256, 0, st>>>(skip, pooled, Cout, H, W);
}

// up block (sorted): bn_b+0, bn_g+1, conv_b+2, conv_w+3, mamba+4..16, up_b+17, up_w+18
// x1 in P (Cin,Hin,Win); result lands in Q (Cout, 2Hin, 2Win)
static void run_up(hipStream_t st, void* const* din, int base, float* P, float* Q,
                   const float* skip, int Cin, int Cout, int Hin, int Win, float* ws)
{
  int Ho = 2 * Hin, Wo = 2 * Win, HWo = Ho * Wo;
  hipMemcpyAsync(Q, skip, sizeof(float) * (size_t)Cout * HWo, hipMemcpyDeviceToDevice, st);
  convt2x2_kernel<<<ceil_div(Cout * HWo, 256), 256, 0, st>>>(
      P, FP(din, base + 18), FP(din, base + 17), Q + (size_t)Cout * HWo, Cin, Cout, Hin, Win);
  conv2d_kernel<<<ceil_div(Cout * HWo, 256), 256, 0, st>>>(
      Q, FP(din, base + 3), FP(din, base + 2), P, Cin, Cout, Ho, Wo, 3, 3, 1);
  run_bn_relu(st, P, FP(din, base + 1), FP(din, base + 0), Cout, HWo, ws + OFF_MV);
  MP p = mp_from(din, base + 4);
  run_mamba(st, p, P, Q, Cout, Ho, Wo, ws);
}

extern "C" void kernel_launch(void* const* d_in, const int* in_sizes, int n_in,
                              void* d_out, int out_size, void* d_ws, size_t ws_size,
                              hipStream_t stream)
{
  (void)in_sizes; (void)n_in; (void)out_size; (void)ws_size;
  float* ws = (float*)d_ws;
  float* BA = ws + OFF_BUFA;
  float* BB = ws + OFF_BUFB;
  const float* x = FP(d_in, I_X);

  // ---- encoder ----
  run_down(stream, d_in, B_ENC1, x,  3,   64, 128, 128, ws, ws + OFF_SKIP1, BA);
  run_down(stream, d_in, B_ENC2, BA, 64,  128, 64, 64,  ws, ws + OFF_SKIP2, BA);
  run_down(stream, d_in, B_ENC3, BA, 128, 256, 32, 32,  ws, ws + OFF_SKIP3, BA);
  run_down(stream, d_in, B_ENC4, BA, 256, 512, 16, 16,  ws, ws + OFF_SKIP4, BA);

  // ---- bottleneck (sorted keys): bn_b+0, bn_g+1, ca_w1+2, ca_w2+3, conv_b+4,
  //      conv_w+5, fus_b+6, fus_bn_b+7, fus_bn_g+8, fus_w+9, mamba+10..22, sa_b+23, sa_w+24
  {
    const int C = 1024, H = 8, W = 8, HW = 64;
    conv2d_kernel<<<ceil_div(C * HW, 256), 256, 0, stream>>>(
        BA, FP(d_in, B_BOT + 5), FP(d_in, B_BOT + 4), BB, 512, C, H, W, 3, 3, 1);
    run_bn_relu(stream, BB, FP(d_in, B_BOT + 1), FP(d_in, B_BOT + 0), C, HW, ws + OFF_MV);
    MP p = mp_from(d_in, B_BOT + 10);
    run_mamba(stream, p, BB, BA, C, H, W, ws);

    float* ST   = ws + OFF_STAT;
    float* avg  = ST;
    float* mx   = ST + 1024;
    float* tmp  = ST + 2048;   // 64
    float* ao   = ST + 2176;   // 1024
    float* mo   = ST + 3200;   // 1024
    float* sain = ST + 4224;   // 2*64
    float* saout= ST + 4352;   // 64
    spatial_mean_kernel<<<C, 256, 0, stream>>>(BA, avg, HW);
    spatial_max_kernel<<<C, 256, 0, stream>>>(BA, mx, HW);
    launch_gemm(stream, ws, avg, C, FP(d_in, B_BOT + 2), nullptr, tmp, 64, 1, 64, C, ACT_RELU);
    launch_gemm(stream, ws, tmp, 64, FP(d_in, B_BOT + 3), nullptr, ao, C, 1, C, 64, ACT_NONE);
    launch_gemm(stream, ws, mx, C, FP(d_in, B_BOT + 2), nullptr, tmp, 64, 1, 64, C, ACT_RELU);
    launch_gemm(stream, ws, tmp, 64, FP(d_in, B_BOT + 3), nullptr, mo, C, 1, C, 64, ACT_NONE);
    ca_scale_kernel<<<ceil_div(C * HW, 256), 256, 0, stream>>>(BA, ao, mo, C, HW);
    chan_meanmax_kernel<<<1, 64, 0, stream>>>(BA, sain, C, HW);
    conv2d_kernel<<<1, 64, 0, stream>>>(sain, FP(d_in, B_BOT + 24), FP(d_in, B_BOT + 23),
                                        saout, 2, 1, H, W, 7, 7, 3);
    sa_scale_kernel<<<ceil_div(C * HW, 256), 256, 0, stream>>>(BA, saout, C, HW);
    conv2d_kernel<<<ceil_div(C * HW, 256), 256, 0, stream>>>(
        BA, FP(d_in, B_BOT + 9), FP(d_in, B_BOT + 6), BB, C, C, H, W, 1, 1, 0);
    run_bn_relu(stream, BB, FP(d_in, B_BOT + 8), FP(d_in, B_BOT + 7), C, HW, ws + OFF_MV);
  }

  // ---- decoder (ping-pong BB <-> BA) ----
  run_up(stream, d_in, B_DEC4, BB, BA, ws + OFF_SKIP4, 1024, 512, 8, 8, ws);   // -> BA
  run_up(stream, d_in, B_DEC3, BA, BB, ws + OFF_SKIP3, 512, 256, 16, 16, ws);  // -> BB
  run_up(stream, d_in, B_DEC2, BB, BA, ws + OFF_SKIP2, 256, 128, 32, 32, ws);  // -> BA
  run_up(stream, d_in, B_DEC1, BA, BB, ws + OFF_SKIP1, 128, 64, 64, 64, ws);   // -> BB

  // ---- final 1x1 conv (64 -> 5) ----
  conv2d_kernel<<<ceil_div(5 * 128 * 128, 256), 256, 0, stream>>>(
      BB, FP(d_in, I_FINAL_W), FP(d_in, I_FINAL_B), (float*)d_out, 64, 5, 128, 128, 1, 1, 0);
}
